// Decoder_39702677684673
// MI455X (gfx1250) — matrix operand order repaired and checked by reading, not compiled, not run on hardware
//
#include <hip/hip_runtime.h>

#define VOCABN 40
#define HDIM   512
#define KSDIM  128
#define VSDIM  128
#define TMAX   1024
#define STEPS  250
#define TXTLEN 251

typedef __attribute__((ext_vector_type(16))) _Float16 v16h;
typedef __attribute__((ext_vector_type(8)))  float    v8f;

union V8 { v8f v; float f[8]; };

__device__ __forceinline__ v8f wmma_f16(v16h a, v16h b, v8f c) {
  // D = A(16x32 f16) * B(32x16 f16) + C(16x16 f32)
  return __builtin_amdgcn_wmma_f32_16x16x32_f16(false, a, false, b, (short)0, c, false, false);
}

// Build an A fragment (16x32 f16) from an LDS row, per ISA 7.12.2:
// lane L: M = L%16; dword j holds K = (j/4)*16 + (L>=16?8:0) + (j%4)*2, +1
__device__ __forceinline__ v16h load_a_frag(const _Float16* rowk, bool hi) {
  union { v16h v; unsigned u[8]; } f;
  const unsigned* p = (const unsigned*)rowk;
  const int off = hi ? 4 : 0;
#pragma unroll
  for (int j = 0; j < 8; ++j)
    f.u[j] = p[(j >> 2) * 8 + off + (j & 3)];
  return f.v;
}

__device__ __forceinline__ float sigf(float x) { return 1.0f / (1.0f + __expf(-x)); }

// Pack W[outN,K] (row-major, used as B = W^T) into per-lane WMMA B fragments:
// tile (nt,kt): lane L holds col n = nt*16 + L%16, K = kt*32 + (j/4)*16 + (L>=16?8:0) + (j%4)*2{,+1}
__global__ void pack_wmma_b(const float* __restrict__ W, _Float16* __restrict__ dst,
                            int outN, int K) {
  const int KT = K >> 5;
  const size_t total = (size_t)outN * K;
  for (size_t i = (size_t)blockIdx.x * blockDim.x + threadIdx.x; i < total;
       i += (size_t)gridDim.x * blockDim.x) {
    int hh    = (int)(i & 15);
    int lanei = (int)((i >> 4) & 31);
    size_t tl = i >> 9;
    int kt = (int)(tl % KT);
    int nt = (int)(tl / KT);
    int j = hh >> 1, r = hh & 1;
    int klocal = (j >> 2) * 16 + ((lanei >= 16) ? 8 : 0) + (j & 3) * 2 + r;
    int n = nt * 16 + (lanei & 15);
    int k = kt * 32 + klocal;
    dst[i] = (_Float16)W[(size_t)n * K + k];
  }
}

__global__ __launch_bounds__(512)
void decoder_persistent(const float* __restrict__ key, const float* __restrict__ values,
                        const float* __restrict__ emb,
                        const float* __restrict__ b1, const float* __restrict__ b2,
                        const float* __restrict__ Wp, const float* __restrict__ bp,
                        const int* __restrict__ lengths, const int* __restrict__ text,
                        const _Float16* __restrict__ w1p, const _Float16* __restrict__ whh1p,
                        const _Float16* __restrict__ w2p, const _Float16* __restrict__ whh2p,
                        float* __restrict__ out) {
  __shared__ float    s_big[16][TMAX];          // attention scores (64 KB)
  __shared__ float    s_ctx[16][VSDIM];         // fp32 context for projection
  __shared__ float    s_h2f32[16][KSDIM];       // fp32 h2 for attention/projection
  __shared__ _Float16 s_inp[16][HDIM + VSDIM];  // [emb | ctx] f16 (A matrix of GEMM1)
  __shared__ _Float16 s_h1f16[16][HDIM];        // h1 f16 (A of GEMM1-hh and GEMM2)
  __shared__ _Float16 s_h2f16[16][KSDIM];       // h2 f16 (A of GEMM2-hh)
  __shared__ int s_tok[16];
  __shared__ int s_len[16];

  const int tid   = threadIdx.x;
  const int wv    = tid >> 5;         // wave id 0..15
  const int lane  = tid & 31;
  const int m16   = lane & 15;
  const bool hi   = lane >= 16;
  const int mbase = hi ? 8 : 0;
  const int wgStart = blockIdx.x * 16;

  const v16h* w1pv   = (const v16h*)w1p;
  const v16h* whh1pv = (const v16h*)whh1p;
  const v16h* w2pv   = (const v16h*)w2p;
  const v16h* whh2pv = (const v16h*)whh2p;

  // --- persistent LSTM cell state in registers ---
  float c1reg[2][8];                  // LSTM1 c: uTiles {wv, wv+16}, rows mbase..mbase+7
  float c2reg[8];                     // LSTM2 c: uTile wv (waves 0..7 only)
#pragma unroll
  for (int u = 0; u < 2; ++u)
#pragma unroll
    for (int r = 0; r < 8; ++r) c1reg[u][r] = 0.f;
#pragma unroll
  for (int r = 0; r < 8; ++r) c2reg[r] = 0.f;

  // --- init LDS state ---
  for (int idx = tid; idx < 16 * HDIM; idx += 512) {
    int row = idx >> 9, u = idx & 511;
    s_h1f16[row][u] = (_Float16)0.f;
  }
  for (int idx = tid; idx < 16 * KSDIM; idx += 512) {
    int row = idx >> 7, u = idx & 127;
    s_h2f32[row][u] = 0.f;
    s_h2f16[row][u] = (_Float16)0.f;
  }
  if (tid < 16) s_len[tid] = lengths[wgStart + tid];
  __syncthreads();

  // --- attention: one wave per batch row (bandwidth-bound; VALU fp32 + prefetch) ---
  auto attend = [&]() {
    const int row = wv;
    const int b = wgStart + row;
    float* sc = &s_big[row][0];
    const float* keyB = key + (size_t)b * TMAX * KSDIM;
    const int len = s_len[row];
    const float4* hp = (const float4*)&s_h2f32[row][0];
#pragma unroll 1
    for (int tp = lane; tp < TMAX; tp += 32) {
      const float4* kp = (const float4*)(keyB + (size_t)tp * KSDIM);
      __builtin_prefetch(keyB + (size_t)(tp + 32) * KSDIM, 0, 0);
      float e = 0.f;
#pragma unroll 8
      for (int kk = 0; kk < 32; ++kk) {
        float4 kv = kp[kk]; float4 hv = hp[kk];
        e += kv.x * hv.x + kv.y * hv.y + kv.z * hv.z + kv.w * hv.w;
      }
      sc[tp] = (tp < len) ? e : -1e9f;
    }
    // softmax within the wave (LDS ops are in-order within a wave)
    float mx = -3.0e38f;
#pragma unroll 1
    for (int tp = lane; tp < TMAX; tp += 32) mx = fmaxf(mx, sc[tp]);
#pragma unroll
    for (int off = 16; off >= 1; off >>= 1) mx = fmaxf(mx, __shfl_xor(mx, off, 32));
    float sum = 0.f;
#pragma unroll 1
    for (int tp = lane; tp < TMAX; tp += 32) {
      float ee = __expf(sc[tp] - mx);
      sc[tp] = ee;
      sum += ee;
    }
#pragma unroll
    for (int off = 16; off >= 1; off >>= 1) sum += __shfl_xor(sum, off, 32);
    const float inv = 1.f / sum;
    const float* valB = values + (size_t)b * TMAX * VSDIM;
    float a0 = 0.f, a1 = 0.f, a2 = 0.f, a3 = 0.f;
#pragma unroll 1
    for (int tp = 0; tp < TMAX; ++tp) {
      float a = sc[tp];
      const float* vp = valB + (size_t)tp * VSDIM;
      __builtin_prefetch(vp + VSDIM, 0, 0);
      a0 += a * vp[lane];      a1 += a * vp[lane + 32];
      a2 += a * vp[lane + 64]; a3 += a * vp[lane + 96];
    }
    const float c0 = a0 * inv, c1 = a1 * inv, c2 = a2 * inv, c3 = a3 * inv;
    s_ctx[row][lane]      = c0; s_ctx[row][lane + 32] = c1;
    s_ctx[row][lane + 64] = c2; s_ctx[row][lane + 96] = c3;
    s_inp[row][HDIM + lane]      = (_Float16)c0; s_inp[row][HDIM + lane + 32] = (_Float16)c1;
    s_inp[row][HDIM + lane + 64] = (_Float16)c2; s_inp[row][HDIM + lane + 96] = (_Float16)c3;
  };

  attend();          // initial ctx = attend(h2 = 0)
  __syncthreads();

#pragma unroll 1
  for (int t = 0; t < STEPS; ++t) {
    if (tid < 16) s_tok[tid] = text[(size_t)(wgStart + tid) * TXTLEN + t];
    __syncthreads();
    // embedding gather -> s_inp[:, 0:512] (emb is L2-hot, 80 KB)
    for (int idx = tid; idx < 16 * HDIM; idx += 512) {
      int row = idx >> 9, u = idx & 511;
      s_inp[row][u] = (_Float16)emb[(size_t)s_tok[row] * HDIM + u];
    }
    __syncthreads();

    // ===== GEMM1 (WMMA): gates1 = inp @ W_ih1^T + h1 @ W_hh1^T =====
    // wave wv owns uTiles {wv, wv+16}; per uTile all 4 gates (i,f,g,o) stay in regs.
    // NOTE: kt loops deliberately kept rolled (#pragma unroll 1) so B fragments
    // stream per-iteration from global -> no VGPR spills.
    V8 acc1[2][4];
#pragma unroll
    for (int u = 0; u < 2; ++u)
#pragma unroll
      for (int g = 0; g < 4; ++g)
#pragma unroll
        for (int r = 0; r < 8; ++r) acc1[u][g].f[r] = 0.f;

#pragma unroll 1
    for (int kt = 0; kt < 20; ++kt) {                    // K = 640 (inp)
      v16h a = load_a_frag(&s_inp[m16][kt * 32], hi);
#pragma unroll
      for (int u = 0; u < 2; ++u) {
        int uTile = wv + u * 16;
#pragma unroll
        for (int g = 0; g < 4; ++g) {
          int nt = g * 32 + uTile;
          v16h bf = w1pv[(size_t)(nt * 20 + kt) * 32 + lane];
          acc1[u][g].v = wmma_f16(a, bf, acc1[u][g].v);
        }
      }
    }
#pragma unroll 1
    for (int kt = 0; kt < 16; ++kt) {                    // K = 512 (h1_prev)
      v16h a = load_a_frag(&s_h1f16[m16][kt * 32], hi);
#pragma unroll
      for (int u = 0; u < 2; ++u) {
        int uTile = wv + u * 16;
#pragma unroll
        for (int g = 0; g < 4; ++g) {
          int nt = g * 32 + uTile;
          v16h bf = whh1pv[(size_t)(nt * 16 + kt) * 32 + lane];
          acc1[u][g].v = wmma_f16(a, bf, acc1[u][g].v);
        }
      }
    }
    __syncthreads();   // everyone done reading old s_h1f16

    // LSTM1 cell, fully in registers; write new h1 (f16) to LDS
#pragma unroll
    for (int u = 0; u < 2; ++u) {
      const int ucol = (wv + u * 16) * 16 + m16;         // 0..511
#pragma unroll
      for (int r = 0; r < 8; ++r) {
        const int M = mbase + r;
        float gi = acc1[u][0].f[r] + b1[ucol];
        float gf = acc1[u][1].f[r] + b1[HDIM + ucol];
        float gg = acc1[u][2].f[r] + b1[2 * HDIM + ucol];
        float go = acc1[u][3].f[r] + b1[3 * HDIM + ucol];
        float c = sigf(gf) * c1reg[u][r] + sigf(gi) * tanhf(gg);
        c1reg[u][r] = c;
        s_h1f16[M][ucol] = (_Float16)(sigf(go) * tanhf(c));
      }
    }
    __syncthreads();

    // ===== GEMM2 (WMMA): gates2 = h1 @ W_ih2^T + h2 @ W_hh2^T  (waves 0..7) =====
    V8 acc2[4];
    if (wv < 8) {
#pragma unroll
      for (int g = 0; g < 4; ++g)
#pragma unroll
        for (int r = 0; r < 8; ++r) acc2[g].f[r] = 0.f;
#pragma unroll 1
      for (int kt = 0; kt < 16; ++kt) {                  // K = 512 (h1)
        v16h a = load_a_frag(&s_h1f16[m16][kt * 32], hi);
#pragma unroll
        for (int g = 0; g < 4; ++g) {
          int nt = g * 8 + wv;
          v16h bf = w2pv[(size_t)(nt * 16 + kt) * 32 + lane];
          acc2[g].v = wmma_f16(a, bf, acc2[g].v);
        }
      }
#pragma unroll 1
      for (int kt = 0; kt < 4; ++kt) {                   // K = 128 (h2_prev)
        v16h a = load_a_frag(&s_h2f16[m16][kt * 32], hi);
#pragma unroll
        for (int g = 0; g < 4; ++g) {
          int nt = g * 8 + wv;
          v16h bf = whh2pv[(size_t)(nt * 4 + kt) * 32 + lane];
          acc2[g].v = wmma_f16(a, bf, acc2[g].v);
        }
      }
    }
    __syncthreads();   // reads of old s_h2f16 complete

    if (wv < 8) {
      const int ucol = wv * 16 + m16;                    // 0..127
#pragma unroll
      for (int r = 0; r < 8; ++r) {
        const int M = mbase + r;
        float gi = acc2[0].f[r] + b2[ucol];
        float gf = acc2[1].f[r] + b2[KSDIM + ucol];
        float gg = acc2[2].f[r] + b2[2 * KSDIM + ucol];
        float go = acc2[3].f[r] + b2[3 * KSDIM + ucol];
        float c = sigf(gf) * c2reg[r] + sigf(gi) * tanhf(gg);
        c2reg[r] = c;
        float h = sigf(go) * tanhf(c);
        s_h2f32[M][ucol] = h;
        s_h2f16[M][ucol] = (_Float16)h;
      }
    }
    __syncthreads();

    // ===== attention (writes s_ctx and next-step ctx half of s_inp) =====
    attend();
    __syncthreads();

    // ===== projection: pred[16,40] = [h2|ctx] @ Wp^T + bp =====
    for (int idx = tid; idx < 16 * VOCABN; idx += 512) {
      int row = idx / VOCABN, v = idx % VOCABN;
      const float* wr = Wp + (size_t)v * (KSDIM + VSDIM);
      float p = bp[v];
#pragma unroll 1
      for (int j = 0; j < KSDIM; ++j) p += wr[j] * s_h2f32[row][j];
#pragma unroll 1
      for (int j = 0; j < VSDIM; ++j) p += wr[KSDIM + j] * s_ctx[row][j];
      out[((size_t)(wgStart + row) * STEPS + t) * VOCABN + v] = p;
    }
    __syncthreads();
  }
}

extern "C" void kernel_launch(void* const* d_in, const int* in_sizes, int n_in,
                              void* d_out, int out_size, void* d_ws, size_t ws_size,
                              hipStream_t stream) {
  (void)in_sizes; (void)n_in; (void)out_size; (void)ws_size;
  const float* key    = (const float*)d_in[0];
  const float* values = (const float*)d_in[1];
  const float* emb    = (const float*)d_in[2];
  const float* W_ih1  = (const float*)d_in[3];
  const float* W_hh1  = (const float*)d_in[4];
  const float* b1     = (const float*)d_in[5];
  const float* W_ih2  = (const float*)d_in[6];
  const float* W_hh2  = (const float*)d_in[7];
  const float* b2     = (const float*)d_in[8];
  const float* Wp     = (const float*)d_in[9];
  const float* bp     = (const float*)d_in[10];
  const int* lengths  = (const int*)d_in[11];
  const int* text     = (const int*)d_in[12];
  float* out = (float*)d_out;

  _Float16* ws    = (_Float16*)d_ws;
  _Float16* w1p   = ws;                                   // 2048*640
  _Float16* whh1p = w1p + (size_t)2048 * 640;             // 2048*512
  _Float16* w2p   = whh1p + (size_t)2048 * 512;           // 512*512
  _Float16* whh2p = w2p + (size_t)512 * 512;              // 512*128

  pack_wmma_b<<<1024, 256, 0, stream>>>(W_ih1, w1p, 2048, 640);
  pack_wmma_b<<<1024, 256, 0, stream>>>(W_hh1, whh1p, 2048, 512);
  pack_wmma_b<<<512,  256, 0, stream>>>(W_ih2, w2p, 512, 512);
  pack_wmma_b<<<256,  256, 0, stream>>>(W_hh2, whh2p, 512, 128);

  decoder_persistent<<<16, 512, 0, stream>>>(key, values, emb, b1, b2, Wp, bp,
                                             lengths, text, w1p, whh1p, w2p, whh2p, out);
}
